// AntisymmetricLayer_54056458387748
// MI455X (gfx1250) — matrix operand order repaired and checked by reading, not compiled, not run on hardware
//
#include <hip/hip_runtime.h>

typedef __attribute__((ext_vector_type(16))) _Float16 v16h;
typedef __attribute__((ext_vector_type(8)))  float    v8f;
typedef __attribute__((ext_vector_type(4)))  float    v4f;
typedef __attribute__((ext_vector_type(4)))  int      v4i;

#define DD   128
#define KK   64
#define RNK  16

// workspace layout (units: _Float16 elements)
#define PF_OFF 0
#define PF_SZ  (KK*4*32*16)          // 64 k * 4 d-chunks * 32 lanes * 16 halfs
#define QF_OFF (PF_SZ)
#define WF_OFF (2*PF_SZ)
#define WF_SZ  (4*4*32*16)
#define WS_HALFS (2*PF_SZ + WF_SZ)

#define PHASE_HALFS (16*4*32*16)     // one 16-k phase of P or Q fragments = 64 KB
#define TR_OFF_B    (4*PHASE_HALFS*2)          // byte offset of transpose region (256 KB)
#define LDS_BYTES   (TR_OFF_B + 8*16*68*4)     // + 34816 B transpose = 296960 B

#if defined(__has_builtin)
#if __has_builtin(__builtin_amdgcn_global_load_async_to_lds_b128)
#define HAVE_ASYNC_LDS 1
#endif
#endif

// operand types for the async-to-LDS builtin (AS1 int4* src, AS3 int4* dst)
typedef __attribute__((address_space(1))) v4i GV4;
typedef __attribute__((address_space(3))) v4i LV4;

// Pre-convert W_lin / P / Q (fp32) into f16 A-matrix fragments in the exact
// per-lane VGPR order of the 16-bit 16x32 WMMA A layout:
//   lane<16 : M=lane,    K order [0..7, 16..23]   (t -> K = t<8 ? t : t+8)
//   lane>=16: M=lane-16, K order [8..15, 24..31]  (+8)
__global__ void prep_frag_kernel(const float* __restrict__ Wl,
                                 const float* __restrict__ P,
                                 const float* __restrict__ Q,
                                 _Float16* __restrict__ ws) {
  int idx = blockIdx.x * blockDim.x + threadIdx.x;
  if (idx >= WS_HALFS) return;
  if (idx < WF_OFF) {                       // P or Q fragments: A[M=r, K=d] = P[k,d,r]
    const float* src = (idx < QF_OFF) ? P : Q;
    int i = (idx < QF_OFF) ? idx : (idx - QF_OFF);
    int t = i & 15, lane = (i >> 4) & 31, c = (i >> 9) & 3, k = i >> 11;
    int d = c * 32 + (t < 8 ? t : t + 8) + ((lane & 16) ? 8 : 0);
    int r = lane & 15;
    ws[idx] = (_Float16)src[(k * DD + d) * RNK + r];
  } else {                                  // W fragments: A[M=k_local, K=d]
    int i = idx - WF_OFF;
    int t = i & 15, lane = (i >> 4) & 31, c = (i >> 9) & 3, kc = i >> 11;
    int d = c * 32 + (t < 8 ? t : t + 8) + ((lane & 16) ? 8 : 0);
    int m = lane & 15;
    ws[idx] = (_Float16)Wl[(kc * 16 + m) * DD + d];
  }
}

__global__ void __launch_bounds__(256)
antisym_wmma_kernel(const float* __restrict__ x1, const float* __restrict__ x2,
                    const _Float16* __restrict__ ws, float* __restrict__ out) {
  extern __shared__ char smem[];
  _Float16* ldsP0 = (_Float16*)smem;               // ping/pong phase buffers
  _Float16* ldsQ0 = ldsP0 + PHASE_HALFS;
  _Float16* ldsP1 = ldsP0 + 2 * PHASE_HALFS;
  _Float16* ldsQ1 = ldsP0 + 3 * PHASE_HALFS;
  float*    ldsT  = (float*)(smem + TR_OFF_B);     // transpose region (separate)

  const int lane = threadIdx.x & 31;
  const int wave = threadIdx.x >> 5;
  const int tile = blockIdx.x * 8 + wave;          // 16 rows per tile
  const int row  = lane & 15;
  const int hi   = lane >> 4;

  const _Float16* Pf = ws + PF_OFF;
  const _Float16* Qf = ws + QF_OFF;
  const _Float16* Wf = ws + WF_OFF;

  // async copy of one 16-k phase of P and Q fragments (64 KB each) into LDS
  auto copy_phase = [&](int ph, _Float16* dP, _Float16* dQ) {
    const _Float16* sP = Pf + ph * PHASE_HALFS;
    const _Float16* sQ = Qf + ph * PHASE_HALFS;
#if defined(HAVE_ASYNC_LDS)
#pragma unroll
    for (int i = 0; i < 16; ++i) {                 // 256 thr * 16 B * 16 = 64 KB
      int off = (threadIdx.x + i * 256) * 8;       // halfs (8 halfs = 16 B)
      __builtin_amdgcn_global_load_async_to_lds_b128((GV4*)(sP + off),
                                                     (LV4*)(dP + off), 0, 0);
      __builtin_amdgcn_global_load_async_to_lds_b128((GV4*)(sQ + off),
                                                     (LV4*)(dQ + off), 0, 0);
    }
#else
#pragma unroll
    for (int i = 0; i < 16; ++i) {
      int idx = threadIdx.x + i * 256;
      ((v4i*)dP)[idx] = ((const v4i*)sP)[idx];
      ((v4i*)dQ)[idx] = ((const v4i*)sQ)[idx];
    }
#endif
  };

  // ---- Build B operands: Z^T, S^T (d x n), 4 chunks of Kdim=32, f16 ----
  // Lane L holds column n=L%16; its 16 halfs are d's [0..7,16..23] (+8 for L>=16) of row n.
  const float* p1 = x1 + (size_t)(tile * 16 + row) * DD + hi * 8;
  const float* p2 = x2 + (size_t)(tile * 16 + row) * DD + hi * 8;
  v16h Bz[4], Bs[4];
#pragma unroll
  for (int c = 0; c < 4; ++c) {
#pragma unroll
    for (int q = 0; q < 4; ++q) {
      v4f a = *(const v4f*)(p1 + c * 32 + q * 4 + (q >= 2 ? 8 : 0));
      v4f b = *(const v4f*)(p2 + c * 32 + q * 4 + (q >= 2 ? 8 : 0));
#pragma unroll
      for (int e = 0; e < 4; ++e) {
        Bz[c][q * 4 + e] = (_Float16)(a[e] - b[e]);
        Bs[c][q * 4 + e] = (_Float16)(a[e] + b[e]);
      }
    }
  }

  // prologue: start phase 0 copy after B operands are built
  copy_phase(0, ldsP0, ldsQ0);

  // ---- 4 phases of 16 k; per phase: lin chunk + bilinear, double-buffered ----
  for (int kc = 0; kc < 4; ++kc) {
    _Float16* bP = (kc & 1) ? ldsP1 : ldsP0;
    _Float16* bQ = (kc & 1) ? ldsQ1 : ldsQ0;
    if (kc < 3) {
      if (kc > 0) __syncthreads();                 // next buffer fully consumed
      copy_phase(kc + 1, (kc & 1) ? ldsP0 : ldsP1, (kc & 1) ? ldsQ0 : ldsQ1);
#if defined(HAVE_ASYNC_LDS)
      // ASYNCcnt is in-order: <=32 outstanding => phase kc's 32 loads landed
      asm volatile("s_wait_asynccnt 0x20" ::: "memory");
#endif
    } else {
#if defined(HAVE_ASYNC_LDS)
      asm volatile("s_wait_asynccnt 0x0" ::: "memory");
#endif
    }
    __syncthreads();                               // staged data visible to WG

    // linear term chunk kc: bacc = W_kc x Z^T, already in (k,n) layout
    v8f bacc = {};
#pragma unroll
    for (int c = 0; c < 4; ++c) {
      v16h aw = *(const v16h*)(Wf + (((kc * 4 + c) * 32 + lane) << 4));
      bacc = __builtin_amdgcn_wmma_f32_16x16x32_f16(false, aw, false, Bz[c],
                                                    (short)0, bacc, false, false);
    }

    // bilinear: per k, zP^T (r x n) and sQ^T via 8 WMMAs from LDS fragments
#pragma unroll
    for (int kk = 0; kk < 16; ++kk) {
      v8f zp = {}, sq = {};
#pragma unroll
      for (int c = 0; c < 4; ++c) {
        v16h ap = *(const v16h*)(bP + (((kk * 4 + c) * 32 + lane) << 4));
        v16h aq = *(const v16h*)(bQ + (((kk * 4 + c) * 32 + lane) << 4));
        zp = __builtin_amdgcn_wmma_f32_16x16x32_f16(false, ap, false, Bz[c],
                                                    (short)0, zp, false, false);
        sq = __builtin_amdgcn_wmma_f32_16x16x32_f16(false, aq, false, Bs[c],
                                                    (short)0, sq, false, false);
      }
      // lanes<16: r=0..7 partials (n=lane); lanes>=16: r=8..15 (n=lane-16)
      float b = 0.f;
#pragma unroll
      for (int v = 0; v < 8; ++v) b = fmaf(zp[v], sq[v], b);
      b += __shfl_xor(b, 16, 32);                  // full sum over r, per half
      bacc[kk & 7] += (((lane < 16) == (kk < 8)) ? b : 0.0f);  // static index
    }

    // deposit this phase's 16 output columns into per-wave transpose slice
    float* L = ldsT + wave * (16 * 68);
#pragma unroll
    for (int v = 0; v < 8; ++v)
      L[row * 68 + kc * 16 + v + hi * 8] = bacc[v];
  }

  // ---- Read back (n,k)-major and store coalesced b128 (same-wave DS order) ----
  asm volatile("s_wait_dscnt 0" ::: "memory");
  const float* L = ldsT + wave * (16 * 68);
  v4f* outv = (v4f*)(out + (size_t)tile * 16 * KK);
#pragma unroll
  for (int q = 0; q < 8; ++q) {
    int i4 = lane + q * 32;                        // 256 float4 = 16 rows x 64 cols
    int r = i4 >> 4, c4 = i4 & 15;
    v4f val = *(const v4f*)(L + r * 68 + c4 * 4);
    outv[r * 16 + c4] = val;
  }
}

extern "C" void kernel_launch(void* const* d_in, const int* in_sizes, int n_in,
                              void* d_out, int out_size, void* d_ws, size_t ws_size,
                              hipStream_t stream) {
  const float* x1 = (const float*)d_in[0];
  const float* x2 = (const float*)d_in[1];
  const float* Wl = (const float*)d_in[2];
  const float* P  = (const float*)d_in[3];
  const float* Q  = (const float*)d_in[4];
  float* out = (float*)d_out;
  _Float16* ws = (_Float16*)d_ws;

  // 1) convert weights to f16 A-fragments in d_ws (~540 KB)
  prep_frag_kernel<<<(WS_HALFS + 255) / 256, 256, 0, stream>>>(Wl, P, Q, ws);

  // 2) main fused kernel: 131072 rows -> 8192 tiles of 16 rows, 8 waves/block,
  //    290 KB dynamic LDS: double-buffered P/Q fragment staging + transpose
  antisym_wmma_kernel<<<1024, 256, LDS_BYTES, stream>>>(x1, x2, ws, out);
}
